// TriangleMultiplicativeModule_5806795784285
// MI455X (gfx1250) — compile-verified, matrix operand order checked by reading, not run on hardware
//
#include <hip/hip_runtime.h>
#include <hip/hip_bf16.h>
#include <stdint.h>

#define B_ 2
#define N_ 256
#define D_ 128
#define H_ 128
#define M_ (B_ * N_ * N_)      // 131072 rows
#define EPS_ 1e-5f

typedef __attribute__((ext_vector_type(16))) _Float16 v16h;
typedef __attribute__((ext_vector_type(8)))  float    v8f;

#if defined(__has_builtin)
#if __has_builtin(__builtin_amdgcn_global_load_async_to_lds_b128) && \
    __has_builtin(__builtin_amdgcn_s_wait_asynccnt)
#define USE_ASYNC_LDS 1
#endif
#endif

#ifdef USE_ASYNC_LDS
typedef int v4i_t __attribute__((vector_size(4 * sizeof(int))));
typedef __attribute__((address_space(1))) v4i_t* as1_v4i_p;
typedef __attribute__((address_space(3))) v4i_t* as3_v4i_p;
#endif

union FragU { v16h v; uint4 q[2]; };

// Load a 16x32 f16 WMMA A/B fragment from a row-major matrix.
// Layout (ISA 7.12.2, 16-bit A 16x32): lane = (kHalf<<4) | m,
// per lane two contiguous 8-half chunks: k in [kh*8, kh*8+8) and [16+kh*8, 16+kh*8+8).
// base must be 16B aligned, stride a multiple of 8 halfs.
__device__ __forceinline__ v16h load_frag(const _Float16* __restrict__ base, int stride) {
    int lane = threadIdx.x & 31;
    int m  = lane & 15;
    int kh = (lane >> 4) & 1;
    const _Float16* p = base + m * stride + kh * 8;
    FragU f;
    f.q[0] = *reinterpret_cast<const uint4*>(p);
    f.q[1] = *reinterpret_cast<const uint4*>(p + 16);
    return f.v;
}

__device__ __forceinline__ float sigmoidf_(float v) {
    return 1.0f / (1.0f + __expf(-v));
}

// 16B global -> LDS copy: async (GLOBAL_LOAD_ASYNC_TO_LDS_B128, ASYNCcnt) when
// available, else synchronous load + ds_store.
__device__ __forceinline__ void cp16_g2l(const _Float16* g, _Float16* l) {
#ifdef USE_ASYNC_LDS
    __builtin_amdgcn_global_load_async_to_lds_b128((as1_v4i_p)g, (as3_v4i_p)l, 0, 0);
#else
    *reinterpret_cast<uint4*>(l) = *reinterpret_cast<const uint4*>(g);
#endif
}

__device__ __forceinline__ void wait_async_all() {
#ifdef USE_ASYNC_LDS
    __builtin_amdgcn_s_wait_asynccnt(0);
#endif
}

// ---------------------------------------------------------------------------
// K0: transpose + convert one 128x128 f32 weight [k][n] -> f16 WT [n][k]
// ---------------------------------------------------------------------------
__global__ void kT(const float* __restrict__ src, _Float16* __restrict__ dst) {
    int n = blockIdx.x;
    int k = threadIdx.x;
    dst[n * 128 + k] = (_Float16)src[k * 128 + n];
}

// ---------------------------------------------------------------------------
// K1: LN1 + 5 projections (WMMA) + mask + sigmoid gating.
// Emits left/right in transposed layout [b, d, j, i] (f16) and out-gate (f16).
// grid = M/16 blocks, 256 threads (8 waves), wave w owns output cols [16w,16w+16).
// ---------------------------------------------------------------------------
__global__ void __launch_bounds__(256)
k1_ln_proj(const float* __restrict__ x,
           const int* __restrict__ src_mask,
           const float* __restrict__ ln1_g, const float* __restrict__ ln1_b,
           const _Float16* __restrict__ WTl,  const _Float16* __restrict__ WTr,
           const _Float16* __restrict__ WTlg, const _Float16* __restrict__ WTrg,
           const _Float16* __restrict__ WTog,
           const float* __restrict__ bl,  const float* __restrict__ br,
           const float* __restrict__ blg, const float* __restrict__ brg,
           const float* __restrict__ bog,
           _Float16* __restrict__ leftT2, _Float16* __restrict__ rightT2,
           _Float16* __restrict__ gate)
{
    __shared__ __align__(16) _Float16 xn[16][144];   // 288B row stride: 16B aligned
    const int row0 = blockIdx.x * 16;
    const int t = threadIdx.x;

    // ---- LayerNorm over D=128: 16 threads per row, 8 cols each ----
    {
        int r = t >> 4, sub = t & 15;
        size_t row = (size_t)row0 + r;
        const float* xp = x + row * D_ + sub * 8;
        float v[8];
        float s = 0.f, ss = 0.f;
        #pragma unroll
        for (int e = 0; e < 8; ++e) { v[e] = xp[e]; s += v[e]; ss += v[e] * v[e]; }
        #pragma unroll
        for (int off = 8; off >= 1; off >>= 1) {
            s  += __shfl_xor(s,  off, 16);
            ss += __shfl_xor(ss, off, 16);
        }
        float mean = s * (1.f / 128.f);
        float var  = ss * (1.f / 128.f) - mean * mean;
        float rstd = rsqrtf(var + EPS_);
        #pragma unroll
        for (int e = 0; e < 8; ++e) {
            int c = sub * 8 + e;
            xn[r][c] = (_Float16)((v[e] - mean) * rstd * ln1_g[c] + ln1_b[c]);
        }
    }
    __syncthreads();

    // ---- 5 projections: 16x16 tile per wave, K=128 in 4 WMMA steps ----
    const int wave = t >> 5;
    const int col0 = wave * 16;
    v8f aL = {}, aR = {}, aLG = {}, aRG = {}, aOG = {};
    #pragma unroll
    for (int kk = 0; kk < 4; ++kk) {
        int k0 = kk * 32;
        v16h af  = load_frag(&xn[0][k0], 144);
        v16h bL  = load_frag(WTl  + col0 * 128 + k0, 128);
        v16h bR  = load_frag(WTr  + col0 * 128 + k0, 128);
        v16h bLG = load_frag(WTlg + col0 * 128 + k0, 128);
        v16h bRG = load_frag(WTrg + col0 * 128 + k0, 128);
        v16h bOG = load_frag(WTog + col0 * 128 + k0, 128);
        aL  = __builtin_amdgcn_wmma_f32_16x16x32_f16(false, af, false, bL,  (short)0, aL,  false, false);
        aR  = __builtin_amdgcn_wmma_f32_16x16x32_f16(false, af, false, bR,  (short)0, aR,  false, false);
        aLG = __builtin_amdgcn_wmma_f32_16x16x32_f16(false, af, false, bLG, (short)0, aLG, false, false);
        aRG = __builtin_amdgcn_wmma_f32_16x16x32_f16(false, af, false, bRG, (short)0, aRG, false, false);
        aOG = __builtin_amdgcn_wmma_f32_16x16x32_f16(false, af, false, bOG, (short)0, aOG, false, false);
    }

    // ---- Epilogue: C/D layout lane=(hi<<4)|n, vgpr r -> M=r+8*hi ----
    const int lane = t & 31;
    const int n = lane & 15, hi = (lane >> 4) & 1;
    const int col = col0 + n;
    #pragma unroll
    for (int r = 0; r < 8; ++r) {
        int m   = r + 8 * hi;
        int row = row0 + m;
        int b = row >> 16;            // N*N = 65536
        int i = (row >> 8) & 255;
        int j = row & 255;
        float mk = (float)(src_mask[b * N_ + i] * src_mask[b * N_ + j]);
        float lv = (aL[r] + bl[col]) * mk * sigmoidf_(aLG[r] + blg[col]);
        float rv = (aR[r] + br[col]) * mk * sigmoidf_(aRG[r] + brg[col]);
        float og = sigmoidf_(aOG[r] + bog[col]);
        // left[b,k=i,j,d] -> leftT2[b][d][j][i]; right[b,k=i,i'=j,d] -> rightT2[b][d][j][i]
        size_t off2 = (((size_t)(b * D_ + col) * N_) + j) * N_ + i;
        leftT2[off2]  = (_Float16)lv;
        rightT2[off2] = (_Float16)rv;
        gate[(size_t)row * H_ + col] = (_Float16)og;
    }
}

// ---------------------------------------------------------------------------
// K2: out[b,i,j,d] = sum_k right[b,k,i,d] * left[b,k,j,d]
//   = rightT2[b,d,i,:] dot leftT2[b,d,j,:]  (both row-major stride N over k)
// 64x64 output tile per block, double-buffered 64x64 LDS panels (async copy),
// 8 waves, 2 accumulators per wave, 16 WMMAs per wave.
// grid = B*D*16 blocks, 256 threads.
// ---------------------------------------------------------------------------
__global__ void __launch_bounds__(256)
k2_einsum(const _Float16* __restrict__ leftT2, const _Float16* __restrict__ rightT2,
          float* __restrict__ mm)
{
    __shared__ __align__(16) _Float16 As[2][64 * 72];  // 72-half stride: 144B, 16B-aligned rows
    __shared__ __align__(16) _Float16 Bs[2][64 * 72];

    const int t    = threadIdx.x;
    const int wave = t >> 5;
    const int bd   = blockIdx.x >> 4;                 // (b*D + d)
    const int tb   = blockIdx.x & 15;
    const int i0   = (tb >> 2) * 64;
    const int j0   = (tb & 3) * 64;
    const int b    = bd >> 7, d = bd & 127;

    const _Float16* Rp = rightT2 + (size_t)bd * (N_ * N_) + (size_t)i0 * N_;  // A: [i][k]
    const _Float16* Lp = leftT2  + (size_t)bd * (N_ * N_) + (size_t)j0 * N_;  // B^T: [j][k]

    // Stage a 64x64 panel: 512 chunks of 8 halfs, 2 per thread.
    auto stage = [&](const _Float16* src, _Float16* dst, int k0) {
        #pragma unroll
        for (int c = 0; c < 2; ++c) {
            int id  = c * 256 + t;
            int row = id >> 3;
            int col = (id & 7) * 8;
            cp16_g2l(src + row * N_ + k0 + col, dst + row * 72 + col);
        }
    };

    const int ti = wave >> 2;   // wave computes tiles (ti, tj) and (ti+2, tj) of 4x4
    const int tj = wave & 3;

    stage(Rp, As[0], 0);
    stage(Lp, Bs[0], 0);
    wait_async_all();
    __syncthreads();

    v8f acc0 = {}, acc1 = {};
    #pragma unroll
    for (int step = 0; step < 4; ++step) {
        const int cur = step & 1;
        if (step < 3) {                               // prefetch next K panel
            stage(Rp, As[cur ^ 1], (step + 1) * 64);
            stage(Lp, Bs[cur ^ 1], (step + 1) * 64);
        }
        #pragma unroll
        for (int kk = 0; kk < 2; ++kk) {
            int k0 = kk * 32;
            v16h a0 = load_frag(&As[cur][(ti * 16) * 72 + k0], 72);
            v16h a1 = load_frag(&As[cur][((ti + 2) * 16) * 72 + k0], 72);
            v16h bf = load_frag(&Bs[cur][(tj * 16) * 72 + k0], 72);
            acc0 = __builtin_amdgcn_wmma_f32_16x16x32_f16(false, a0, false, bf, (short)0, acc0, false, false);
            acc1 = __builtin_amdgcn_wmma_f32_16x16x32_f16(false, a1, false, bf, (short)0, acc1, false, false);
        }
        wait_async_all();
        __syncthreads();
    }

    const int lane = t & 31;
    const int n = lane & 15, hi = (lane >> 4) & 1;
    #pragma unroll
    for (int r = 0; r < 8; ++r) {
        int ia = i0 + ti * 16 + r + 8 * hi;
        int ib = i0 + (ti + 2) * 16 + r + 8 * hi;
        int j  = j0 + tj * 16 + n;
        mm[(((size_t)(b * N_ + ia) * N_) + j) * H_ + d] = acc0[r];
        mm[(((size_t)(b * N_ + ib) * N_) + j) * H_ + d] = acc1[r];
    }
}

// ---------------------------------------------------------------------------
// K3: LN2(mm) * out_gate -> WMMA GEMM with Wo^T -> + bo -> d_out (f32).
// ---------------------------------------------------------------------------
__global__ void __launch_bounds__(256)
k3_ln2_out(const float* __restrict__ mm, const _Float16* __restrict__ gate,
           const float* __restrict__ ln2_g, const float* __restrict__ ln2_b,
           const _Float16* __restrict__ WTo, const float* __restrict__ bo,
           float* __restrict__ out)
{
    __shared__ __align__(16) _Float16 yn[16][144];
    const int row0 = blockIdx.x * 16;
    const int t = threadIdx.x;

    {
        int r = t >> 4, sub = t & 15;
        size_t row = (size_t)row0 + r;
        const float* xp = mm + row * H_ + sub * 8;
        const _Float16* gp = gate + row * H_ + sub * 8;
        float v[8];
        float s = 0.f, ss = 0.f;
        #pragma unroll
        for (int e = 0; e < 8; ++e) { v[e] = xp[e]; s += v[e]; ss += v[e] * v[e]; }
        #pragma unroll
        for (int off = 8; off >= 1; off >>= 1) {
            s  += __shfl_xor(s,  off, 16);
            ss += __shfl_xor(ss, off, 16);
        }
        float mean = s * (1.f / 128.f);
        float var  = ss * (1.f / 128.f) - mean * mean;
        float rstd = rsqrtf(var + EPS_);
        #pragma unroll
        for (int e = 0; e < 8; ++e) {
            int c = sub * 8 + e;
            float y = ((v[e] - mean) * rstd * ln2_g[c] + ln2_b[c]) * (float)gp[e];
            yn[r][c] = (_Float16)y;
        }
    }
    __syncthreads();

    const int wave = t >> 5;
    const int col0 = wave * 16;
    v8f acc = {};
    #pragma unroll
    for (int kk = 0; kk < 4; ++kk) {
        int k0 = kk * 32;
        v16h af = load_frag(&yn[0][k0], 144);
        v16h bf = load_frag(WTo + col0 * 128 + k0, 128);
        acc = __builtin_amdgcn_wmma_f32_16x16x32_f16(false, af, false, bf, (short)0, acc, false, false);
    }

    const int lane = t & 31;
    const int n = lane & 15, hi = (lane >> 4) & 1;
    const int col = col0 + n;
    #pragma unroll
    for (int r = 0; r < 8; ++r) {
        int row = row0 + r + 8 * hi;
        out[(size_t)row * D_ + col] = acc[r] + bo[col];
    }
}

// ---------------------------------------------------------------------------
// Workspace layout (bytes):
//   [0, 196608)                  : 6 f16 transposed weights (6 * 128*128*2)
//   [196608, +32MiB)             : leftT2  f16 [B,D,N,N]
//   [.., +32MiB)                 : rightT2 f16 [B,D,N,N]
//   [.., +32MiB)                 : gate    f16 [B,N,N,H]
//   [.., +64MiB)                 : mm      f32 [B,N,N,H]
// total ~160 MiB
// ---------------------------------------------------------------------------
extern "C" void kernel_launch(void* const* d_in, const int* in_sizes, int n_in,
                              void* d_out, int out_size, void* d_ws, size_t ws_size,
                              hipStream_t stream) {
    (void)in_sizes; (void)n_in; (void)out_size; (void)ws_size;
    const float* x        = (const float*)d_in[0];
    const int*   src_mask = (const int*)  d_in[1];
    const float* ln1_g    = (const float*)d_in[2];
    const float* ln1_b    = (const float*)d_in[3];
    const float* Wl       = (const float*)d_in[4];
    const float* bl       = (const float*)d_in[5];
    const float* Wr       = (const float*)d_in[6];
    const float* br       = (const float*)d_in[7];
    const float* Wlg      = (const float*)d_in[8];
    const float* blg      = (const float*)d_in[9];
    const float* Wrg      = (const float*)d_in[10];
    const float* brg      = (const float*)d_in[11];
    const float* Wog      = (const float*)d_in[12];
    const float* bog      = (const float*)d_in[13];
    const float* ln2_g    = (const float*)d_in[14];
    const float* ln2_b    = (const float*)d_in[15];
    const float* Wo       = (const float*)d_in[16];
    const float* bo       = (const float*)d_in[17];

    char* ws = (char*)d_ws;
    _Float16* WT      = (_Float16*)ws;            // 6 * 16384 halfs
    _Float16* WTl     = WT + 0 * 16384;
    _Float16* WTr     = WT + 1 * 16384;
    _Float16* WTlg    = WT + 2 * 16384;
    _Float16* WTrg    = WT + 3 * 16384;
    _Float16* WTog    = WT + 4 * 16384;
    _Float16* WTo     = WT + 5 * 16384;
    _Float16* leftT2  = (_Float16*)(ws + 196608);
    _Float16* rightT2 = leftT2 + (size_t)16777216;
    _Float16* gate    = rightT2 + (size_t)16777216;
    float*    mm      = (float*)(ws + 196608 + (size_t)3 * 16777216 * 2);

    kT<<<128, 128, 0, stream>>>(Wl,  WTl);
    kT<<<128, 128, 0, stream>>>(Wr,  WTr);
    kT<<<128, 128, 0, stream>>>(Wlg, WTlg);
    kT<<<128, 128, 0, stream>>>(Wrg, WTrg);
    kT<<<128, 128, 0, stream>>>(Wog, WTog);
    kT<<<128, 128, 0, stream>>>(Wo,  WTo);

    k1_ln_proj<<<M_ / 16, 256, 0, stream>>>(x, src_mask, ln1_g, ln1_b,
                                            WTl, WTr, WTlg, WTrg, WTog,
                                            bl, br, blg, brg, bog,
                                            leftT2, rightT2, gate);

    k2_einsum<<<B_ * D_ * 16, 256, 0, stream>>>(leftT2, rightT2, mm);

    k3_ln2_out<<<M_ / 16, 256, 0, stream>>>(mm, gate, ln2_g, ln2_b, WTo, bo,
                                            (float*)d_out);
}
